// ResonanceMemoryAttentionV15_73615739453972
// MI455X (gfx1250) — matrix operand order repaired and checked by reading, not compiled, not run on hardware
//
#include <hip/hip_runtime.h>
#include <hip/hip_bf16.h>
#include <math.h>

// ---------------------------------------------------------------------------
// ResonanceMemoryAttention for MI455X (gfx1250), wave32 + WMMA f16 path.
// B=4, L=512, D=768, H=12, DH=64, R=128.  Chunked linear attention, C=64.
// GEMM tiles staged by the Tensor Data Mover (tensor_load_to_lds), double
// buffered and synchronized with s_wait_tensorcnt.
// ---------------------------------------------------------------------------

typedef _Float16 h16;
typedef __attribute__((ext_vector_type(16))) _Float16 v16h;
typedef __attribute__((ext_vector_type(8)))  float    v8f;
typedef uint32_t u32x4 __attribute__((ext_vector_type(4)));
typedef uint32_t u32x8 __attribute__((ext_vector_type(8)));

#define PI_F 3.14159265358979323846f

#define WMMA_F16(a, b, c) \
  __builtin_amdgcn_wmma_f32_16x16x32_f16(false, (a), false, (b), (short)0, (c), false, false)

// ---------------------------------------------------------------------------
// TDM 2D tile load: global (row-major, stride in elements) -> LDS, f16 data.
// Builds a D# per CDNA5 ISA ch.8 (groups 0/1 only: 2D tensor) and issues
// tensor_load_to_lds. OOB reads beyond tensor_d0/d1 are zero-filled by HW.
// Caller must be wave-uniform; completion via s_wait_tensorcnt.
// ---------------------------------------------------------------------------
__device__ __forceinline__ void tdm_load_2d(const void* lds_dst, const void* gsrc,
                                            uint32_t tensor_d0, uint32_t tensor_d1,
                                            uint32_t tile_d0, uint32_t tile_d1,
                                            uint32_t stride_elems) {
  uint64_t ga = (uint64_t)(uintptr_t)gsrc;
  u32x4 g0;
  g0[0] = 1u;                                         // count=1, user desc
  g0[1] = (uint32_t)(uintptr_t)lds_dst;               // lds_addr (bytes)
  g0[2] = (uint32_t)ga;                               // global_addr[31:0]
  g0[3] = (uint32_t)((ga >> 32) & 0x1FFFFFFu)         // global_addr[56:32]
        | (2u << 30);                                 // type = 2 ("image")
  u32x8 g1;
  g1[0] = (1u << 16);                                 // data_size = 2 bytes
  g1[1] = (tensor_d0 & 0xFFFFu) << 16;                // tensor_dim0[15:0]
  g1[2] = ((tensor_d0 >> 16) & 0xFFFFu)               // tensor_dim0[31:16]
        | ((tensor_d1 & 0xFFFFu) << 16);              // tensor_dim1[15:0]
  g1[3] = ((tensor_d1 >> 16) & 0xFFFFu)               // tensor_dim1[31:16]
        | (tile_d0 << 16);                            // tile_dim0
  g1[4] = tile_d1 & 0xFFFFu;                          // tile_dim1 (tile_dim2=0)
  g1[5] = stride_elems;                               // tensor_dim0_stride[31:0]
  g1[6] = 0u;                                         // stride hi / dim1 stride
  g1[7] = 0u;
  asm volatile("tensor_load_to_lds %0, %1" :: "s"(g0), "s"(g1) : "memory");
}

// ---------------------------------------------------------------------------
// WMMA fragment loaders (per CDNA5 ISA 7.12.2 VGPR layouts, wave32).
// ---------------------------------------------------------------------------

// A[m][k] = src[m*stride + k]
__device__ __forceinline__ v16h load_a_frag(const h16* src, int stride) {
  int lane = threadIdx.x & 31;
  int m  = lane & 15;
  int k8 = ((lane >> 4) & 1) * 8;
  v16h a;
#pragma unroll
  for (int g = 0; g < 2; ++g)
#pragma unroll
    for (int i = 0; i < 8; ++i)
      a[g * 8 + i] = src[m * stride + g * 16 + k8 + i];
  return a;
}

// A[m][k] = srcT[k*stride + m]   (transposed source, used for Kg^T @ V)
__device__ __forceinline__ v16h load_aT_frag(const h16* srcT, int stride) {
  int lane = threadIdx.x & 31;
  int m  = lane & 15;
  int k8 = ((lane >> 4) & 1) * 8;
  v16h a;
#pragma unroll
  for (int g = 0; g < 2; ++g)
#pragma unroll
    for (int i = 0; i < 8; ++i)
      a[g * 8 + i] = srcT[(g * 16 + k8 + i) * stride + m];
  return a;
}

// B[k][n] = src[k*stride + n]
__device__ __forceinline__ v16h load_b_frag(const h16* src, int stride) {
  int lane = threadIdx.x & 31;
  int n   = lane & 15;
  int k16 = ((lane >> 4) & 1) * 8;
  v16h b;
#pragma unroll
  for (int h = 0; h < 16; ++h)
    b[h] = src[((h >> 3) * 16 + k16 + (h & 7)) * stride + n];
  return b;
}

// B[k][n] = srcT[n*stride + k]   (transposed source, contiguous: Q @ Kg^T)
__device__ __forceinline__ v16h load_bT_frag(const h16* srcT, int stride) {
  int lane = threadIdx.x & 31;
  int n   = lane & 15;
  int k16 = ((lane >> 4) & 1) * 8;
  v16h b;
#pragma unroll
  for (int h = 0; h < 16; ++h)
    b[h] = srcT[n * stride + (h >> 3) * 16 + k16 + (h & 7)];
  return b;
}

__device__ __forceinline__ float elup(float x) {   // elu(x)+1
  return x > 0.f ? x + 1.f : __expf(x);
}
__device__ __forceinline__ float sigm(float x) {
  return 1.f / (1.f + __expf(-x));
}

// ---------------------------------------------------------------------------
// f32 -> f16 conversion
// ---------------------------------------------------------------------------
__global__ void cvt_f16_kernel(const float* __restrict__ src, h16* __restrict__ dst, int n) {
  int i = blockIdx.x * 256 + threadIdx.x;
  if (i < n) dst[i] = (h16)src[i];
}

// ---------------------------------------------------------------------------
// LayerNorm over D=768; writes raw-x f16 and x_norm f16. One 64-thread block
// per (b,l) row.
// ---------------------------------------------------------------------------
__global__ __launch_bounds__(64)
void layernorm_kernel(const float* __restrict__ x, const float* __restrict__ g,
                      const float* __restrict__ bb, h16* __restrict__ xh,
                      h16* __restrict__ xnh) {
  const int row = blockIdx.x;
  const float* xr = x + (size_t)row * 768;
  __shared__ float red[64];
  int t = threadIdx.x;
  float s = 0.f;
  for (int i = t; i < 768; i += 64) s += xr[i];
  red[t] = s; __syncthreads();
#pragma unroll
  for (int off = 32; off > 0; off >>= 1) { if (t < off) red[t] += red[t + off]; __syncthreads(); }
  float mu = red[0] * (1.f / 768.f);
  __syncthreads();
  float v = 0.f;
  for (int i = t; i < 768; i += 64) { float d = xr[i] - mu; v += d * d; }
  red[t] = v; __syncthreads();
#pragma unroll
  for (int off = 32; off > 0; off >>= 1) { if (t < off) red[t] += red[t + off]; __syncthreads(); }
  float rstd = rsqrtf(red[0] * (1.f / 768.f) + 1e-5f);
  for (int i = t; i < 768; i += 64) {
    float xv = xr[i];
    xh [(size_t)row * 768 + i] = (h16)xv;
    xnh[(size_t)row * 768 + i] = (h16)((xv - mu) * rstd * g[i] + bb[i]);
  }
}

// ---------------------------------------------------------------------------
// Generic WMMA GEMM: C = epilogue(A[MxK] @ B[KxN] + bias)
// 128 threads = 4 waves. Tile 64x64; wave w owns row-strip w*16, 4 N-tiles.
// A/B tiles DMA'd into double-buffered LDS by the TDM (wave 0 issues),
// overlapped with WMMA on the other buffer; sync via s_wait_tensorcnt.
// MODE 0: +bias f16 | 1: silu f16 | 2: f32 | 3: +bias f32.
// Requires M%64==0, K%32==0; N guarded (TDM zero-fills OOB columns).
// ---------------------------------------------------------------------------
template <int MODE>
__global__ __launch_bounds__(128)
void gemm_wmma(const h16* __restrict__ A, const h16* __restrict__ B,
               const float* __restrict__ bias, void* __restrict__ Cout,
               int M, int N, int K, int lda, int ldb, int ldc) {
  __shared__ h16 As[2][64 * 32];
  __shared__ h16 Bs[2][32 * 64];
  const int m0 = blockIdx.y * 64;
  const int n0 = blockIdx.x * 64;
  const int tid = threadIdx.x;
  const int wave = tid >> 5;

  const v8f vzero = {0.f, 0.f, 0.f, 0.f, 0.f, 0.f, 0.f, 0.f};
  v8f acc[4] = {vzero, vzero, vzero, vzero};

  // issue TDM descriptors for K-slice k0 into buffer `buf` (wave-uniform)
  auto issue_tiles = [&](int buf, int k0) {
    // A tile: 64 rows x 32 cols, row stride lda
    tdm_load_2d(&As[buf][0], A + (size_t)m0 * lda + k0,
                (uint32_t)(K - k0), (uint32_t)(M - m0), 32u, 64u, (uint32_t)lda);
    // B tile: 32 rows x 64 cols, row stride ldb (OOB cols zero-filled)
    tdm_load_2d(&Bs[buf][0], B + (size_t)k0 * ldb + n0,
                (uint32_t)(N - n0), (uint32_t)(K - k0), 64u, 32u, (uint32_t)ldb);
  };

  if (wave == 0) {
    issue_tiles(0, 0);
    __builtin_amdgcn_s_wait_tensorcnt(0);
  }
  __syncthreads();

  int cur = 0;
  for (int k0 = 0; k0 < K; k0 += 32) {
    if (wave == 0 && k0 + 32 < K) issue_tiles(cur ^ 1, k0 + 32);

    v16h a = load_a_frag(&As[cur][(wave * 16) * 32], 32);
#pragma unroll
    for (int t = 0; t < 4; ++t) {
      v16h b = load_b_frag(&Bs[cur][t * 16], 64);
      acc[t] = WMMA_F16(a, b, acc[t]);
    }

    if (wave == 0 && k0 + 32 < K) __builtin_amdgcn_s_wait_tensorcnt(0);
    __syncthreads();
    cur ^= 1;
  }

  const int lane = tid & 31;
  const int mbase = m0 + wave * 16 + ((lane >> 4) & 1) * 8;
  const int nc = lane & 15;
#pragma unroll
  for (int t = 0; t < 4; ++t) {
    int n = n0 + t * 16 + nc;
    if (n >= N) continue;
    float bv = (MODE == 0 || MODE == 3) ? bias[n] : 0.f;
#pragma unroll
    for (int r = 0; r < 8; ++r) {
      int m = mbase + r;
      float v = acc[t][r] + bv;
      if (MODE == 1) v = v * sigm(v);             // silu
      if (MODE == 0 || MODE == 1) ((h16*)Cout)[(size_t)m * ldc + n] = (h16)v;
      else                        ((float*)Cout)[(size_t)m * ldc + n] = v;
    }
  }
}

// ---------------------------------------------------------------------------
// Gate: params(2048 x 64, f32, 5 per head, [0..3] used) -> gate (B*L*H)
// ---------------------------------------------------------------------------
__global__ void gate_kernel(const float* __restrict__ params,
                            const float* __restrict__ temperature,
                            float* __restrict__ gate_ws,
                            float* __restrict__ gate_out, int n) {
  int i = blockIdx.x * 256 + threadIdx.x;
  if (i >= n) return;
  int row = i / 12, h = i % 12;
  const float* p = params + (size_t)row * 64 + h * 5;
  float sa = sigm(p[0]);
  float sp = tanhf(p[1]) * PI_F;
  float ca = sigm(p[2]);
  float cp = tanhf(p[3]) * PI_F;
  float tmp = fminf(fmaxf(temperature[0], 0.1f), 2.0f);
  float inter = tanhf(sa * ca * __cosf(sp - cp)) * tmp;
  float gv = sigm(inter);
  gate_ws[i]  = gv;
  gate_out[i] = gv;
}

// ---------------------------------------------------------------------------
// Chunked gated linear attention. One block per (b,h): 48 blocks, 256 thr.
// Per chunk C=64:  O = Q@S_prev + tril(Q@Kg^T)@V ;  S += Kg^T@V ;
// den = q . (z_prev + cumsum k_f) ; per-row LayerNorm(DH) -> core f16.
// ---------------------------------------------------------------------------
__global__ __launch_bounds__(256)
void attention_kernel(const h16* __restrict__ qkvh, const float* __restrict__ gate,
                      const float* __restrict__ mn_g, const float* __restrict__ mn_b,
                      h16* __restrict__ core) {
  __shared__ h16 Qs[64 * 64];    // q_f
  __shared__ h16 Kg[64 * 64];    // (1+gate) * k_f
  __shared__ h16 Vs[64 * 64];
  __shared__ h16 Sh[64 * 64];    // S_prev (f16 view)
  __shared__ h16 Ah[64 * 64];    // masked scores
  __shared__ float fbuf[64 * 64];// z-cumsum, then O numerator
  __shared__ float zprev[64];
  __shared__ float denb[64];

  const int b = blockIdx.x / 12;
  const int h = blockIdx.x % 12;
  const int tid = threadIdx.x;
  const int lane = tid & 31;
  const int wave = tid >> 5;
  const int mi = wave >> 1;                 // output tile row 0..3
  const int hi8 = ((lane >> 4) & 1) * 8;
  const int nc = lane & 15;

  const v8f vzero = {0.f, 0.f, 0.f, 0.f, 0.f, 0.f, 0.f, 0.f};
  v8f sacc[2] = {vzero, vzero};             // persistent S state (2 tiles/wave)

  if (tid < 64) zprev[tid] = 0.f;

  for (int chunk = 0; chunk < 8; ++chunk) {
    const int l0 = chunk * 64;
    __syncthreads();   // previous chunk fully consumed Sh / fbuf / tiles

    // ---- 1: publish S_prev (f16) from persistent accumulators ----
#pragma unroll
    for (int t = 0; t < 2; ++t) {
      int ni = (wave & 1) * 2 + t;
#pragma unroll
      for (int r = 0; r < 8; ++r)
        Sh[(mi * 16 + r + hi8) * 64 + ni * 16 + nc] = (h16)sacc[t][r];
    }

    // ---- 2: build Q, Kg, V tiles (elu+1, gate scaling) ----
    {
      int r = tid >> 2;            // chunk row 0..63
      int c0 = (tid & 3) * 16;
      int gl = (b * 512 + l0 + r);
      const h16* src = qkvh + (size_t)gl * 2304 + h * 64;
      float gv = gate[(size_t)gl * 12 + h];
#pragma unroll
      for (int i = 0; i < 16; ++i) {
        float qv = (float)src[c0 + i];
        float kv = (float)src[768 + c0 + i];
        Qs[r * 64 + c0 + i] = (h16)elup(qv);
        Kg[r * 64 + c0 + i] = (h16)(elup(kv) * (1.f + gv));
        Vs[r * 64 + c0 + i] = src[1536 + c0 + i];
      }
    }
    __syncthreads();

    // ---- 3: z column scan (thread d), then denominator (thread l) ----
    if (tid < 64) {
      int d = tid;
      float run = zprev[d];
      const h16* kc = qkvh + (size_t)(b * 512 + l0) * 2304 + 768 + h * 64 + d;
      for (int l = 0; l < 64; ++l) {
        run += elup((float)kc[(size_t)l * 2304]);
        fbuf[l * 64 + d] = run;
      }
      zprev[d] = run;
    }
    __syncthreads();
    if (tid < 64) {
      int l = tid;
      float s = 1e-6f;
      for (int d = 0; d < 64; ++d)
        s += (float)Qs[l * 64 + d] * fbuf[l * 64 + d];
      denb[l] = s;
    }

    // ---- 4: scores = Q @ Kg^T, causal mask, -> Ah ----
#pragma unroll
    for (int t = 0; t < 2; ++t) {
      int ni = (wave & 1) * 2 + t;
      v8f sc = vzero;
#pragma unroll
      for (int ks = 0; ks < 2; ++ks) {
        v16h a = load_a_frag(&Qs[(mi * 16) * 64 + ks * 32], 64);
        v16h bf = load_bT_frag(&Kg[(ni * 16) * 64 + ks * 32], 64);
        sc = WMMA_F16(a, bf, sc);
      }
      int col = ni * 16 + nc;
#pragma unroll
      for (int r = 0; r < 8; ++r) {
        int row = mi * 16 + r + hi8;
        Ah[row * 64 + col] = (h16)((col <= row) ? sc[r] : 0.f);
      }
    }
    __syncthreads();   // Ah ready; den reads of fbuf done -> fbuf reusable as O

    // ---- 5: O = Q @ S_prev + Ah @ V ;  6: S += Kg^T @ V ----
#pragma unroll
    for (int t = 0; t < 2; ++t) {
      int ni = (wave & 1) * 2 + t;
      v8f o = vzero;
#pragma unroll
      for (int ks = 0; ks < 2; ++ks) {
        v16h a = load_a_frag(&Qs[(mi * 16) * 64 + ks * 32], 64);
        v16h bf = load_b_frag(&Sh[(ks * 32) * 64 + ni * 16], 64);
        o = WMMA_F16(a, bf, o);
      }
#pragma unroll
      for (int ks = 0; ks < 2; ++ks) {
        v16h a = load_a_frag(&Ah[(mi * 16) * 64 + ks * 32], 64);
        v16h bf = load_b_frag(&Vs[(ks * 32) * 64 + ni * 16], 64);
        o = WMMA_F16(a, bf, o);
      }
      int col = ni * 16 + nc;
#pragma unroll
      for (int r = 0; r < 8; ++r)
        fbuf[(mi * 16 + r + hi8) * 64 + col] = o[r];

#pragma unroll
      for (int ks = 0; ks < 2; ++ks) {
        v16h a = load_aT_frag(&Kg[(ks * 32) * 64 + mi * 16], 64);
        v16h bf = load_b_frag(&Vs[(ks * 32) * 64 + ni * 16], 64);
        sacc[t] = WMMA_F16(a, bf, sacc[t]);
      }
    }
    __syncthreads();

    // ---- 7: epilogue: divide by den, head-LayerNorm(DH=64), store f16 ----
    if (tid < 64) {
      int l = tid;
      float inv = 1.f / denb[l];
      float mu = 0.f;
      for (int d = 0; d < 64; ++d) mu += fbuf[l * 64 + d] * inv;
      mu *= (1.f / 64.f);
      float var = 0.f;
      for (int d = 0; d < 64; ++d) {
        float u = fbuf[l * 64 + d] * inv - mu;
        var += u * u;
      }
      float rstd = rsqrtf(var * (1.f / 64.f) + 1e-5f);
      h16* dst = core + (size_t)(b * 512 + l0 + l) * 768 + h * 64;
      for (int d = 0; d < 64; ++d) {
        float u = fbuf[l * 64 + d] * inv;
        dst[d] = (h16)((u - mu) * rstd * mn_g[d] + mn_b[d]);
      }
    }
  }
}

// ---------------------------------------------------------------------------
// Host orchestration
// ---------------------------------------------------------------------------
extern "C" void kernel_launch(void* const* d_in, const int* in_sizes, int n_in,
                              void* d_out, int out_size, void* d_ws, size_t ws_size,
                              hipStream_t stream) {
  (void)in_sizes; (void)n_in; (void)out_size; (void)ws_size;
  const float* x    = (const float*)d_in[0];
  const float* Wqkv = (const float*)d_in[1];
  const float* bqkv = (const float*)d_in[2];
  const float* Wb1  = (const float*)d_in[3];
  const float* Wb2  = (const float*)d_in[4];
  const float* temp = (const float*)d_in[5];
  const float* Wproj= (const float*)d_in[6];
  const float* bproj= (const float*)d_in[7];
  const float* ln_g = (const float*)d_in[8];
  const float* ln_b = (const float*)d_in[9];
  const float* mn_g = (const float*)d_in[10];
  const float* mn_b = (const float*)d_in[11];

  const int BL = 4 * 512;              // 2048 rows
  char* ws = (char*)d_ws;
  size_t off = 0;
  auto carve = [&](size_t bytes) -> void* {
    void* p = ws + off;
    off = (off + bytes + 255) & ~(size_t)255;
    return p;
  };
  h16*   xh     = (h16*)carve((size_t)BL * 768 * 2);
  h16*   xnh    = (h16*)carve((size_t)BL * 768 * 2);
  h16*   Wqkvh  = (h16*)carve((size_t)768 * 2304 * 2);
  h16*   Wb1h   = (h16*)carve((size_t)768 * 128 * 2);
  h16*   Wb2h   = (h16*)carve((size_t)128 * 60 * 2);
  h16*   Wprojh = (h16*)carve((size_t)768 * 768 * 2);
  h16*   qkvh   = (h16*)carve((size_t)BL * 2304 * 2);
  h16*   hh     = (h16*)carve((size_t)BL * 128 * 2);
  float* params = (float*)carve((size_t)BL * 64 * 4);
  float* gatews = (float*)carve((size_t)BL * 12 * 4);
  h16*   coreh  = (h16*)carve((size_t)BL * 768 * 2);

  float* out_attn = (float*)d_out;                     // (B,L,D)
  float* out_gate = (float*)d_out + (size_t)BL * 768;  // (B,L,H)

  // weight conversions
  cvt_f16_kernel<<<(768 * 2304 + 255) / 256, 256, 0, stream>>>(Wqkv, Wqkvh, 768 * 2304);
  cvt_f16_kernel<<<(768 * 128  + 255) / 256, 256, 0, stream>>>(Wb1,  Wb1h,  768 * 128);
  cvt_f16_kernel<<<(128 * 60   + 255) / 256, 256, 0, stream>>>(Wb2,  Wb2h,  128 * 60);
  cvt_f16_kernel<<<(768 * 768  + 255) / 256, 256, 0, stream>>>(Wproj,Wprojh,768 * 768);

  // layernorm + f16 copies of x
  layernorm_kernel<<<BL, 64, 0, stream>>>(x, ln_g, ln_b, xh, xnh);

  // qkv = x_norm @ Wqkv + bqkv   (2048 x 2304 x 768) -> f16
  gemm_wmma<0><<<dim3(2304 / 64, BL / 64), 128, 0, stream>>>(
      xnh, Wqkvh, bqkv, qkvh, BL, 2304, 768, 768, 2304, 2304);

  // hh = silu(x @ Wb1)           (2048 x 128 x 768) -> f16
  gemm_wmma<1><<<dim3(128 / 64, BL / 64), 128, 0, stream>>>(
      xh, Wb1h, nullptr, hh, BL, 128, 768, 768, 128, 128);

  // params = hh @ Wb2            (2048 x 60 x 128) -> f32, row stride 64
  gemm_wmma<2><<<dim3(1, BL / 64), 128, 0, stream>>>(
      hh, Wb2h, nullptr, params, BL, 60, 128, 128, 60, 64);

  // gate (to ws + second output)
  gate_kernel<<<(BL * 12 + 255) / 256, 256, 0, stream>>>(
      params, temp, gatews, out_gate, BL * 12);

  // chunked gated linear attention + head layernorm -> core f16
  attention_kernel<<<48, 256, 0, stream>>>(qkvh, gatews, mn_g, mn_b, coreh);

  // out = core @ Wproj + bproj   (2048 x 768 x 768) -> f32 d_out
  gemm_wmma<3><<<dim3(768 / 64, BL / 64), 128, 0, stream>>>(
      coreh, Wprojh, bproj, out_attn, BL, 768, 768, 768, 768, 768);
}
